// MultiHeadedAttention_84293028151875
// MI455X (gfx1250) — matrix operand order repaired and checked by reading, not compiled, not run on hardware
//
#include <hip/hip_runtime.h>
#include <hip/hip_bf16.h>

// ---------------------------------------------------------------------------
// MHA forward for MI455X (gfx1250, wave32, WMMA 16x16x32 bf16 -> f32 acc)
//   B=4, S=2048, D_MODEL=1024, H=16, DK=64
// Pipeline:
//   1) fp32 -> bf16 conversion of x / W
//   2) WMMA GEMM  Y = X @ W^T + b   (W tile staged in LDS via async copies)
//   3) flash-attention (K/V chunks staged in LDS via async copies,
//      online softmax, scores never leave VGPRs)
//   4) WMMA GEMM  out = Attn @ Wo^T + bo  (fp32 out)
// ---------------------------------------------------------------------------

#define S_LEN   2048
#define DMODEL  1024
#define NHEAD   16
#define NBATCH  4
#define DK      64
#define MROWS   (NBATCH * S_LEN)   // 8192

typedef __attribute__((ext_vector_type(16))) __bf16 v16bf;
typedef __attribute__((ext_vector_type(8)))  __bf16 v8bf;
typedef __attribute__((ext_vector_type(8)))  float  v8f;
typedef __attribute__((ext_vector_type(4)))  float  v4f;

union ABFrag { v16bf v; v8bf h[2]; };

// Output modes for the shared GEMM kernel
#define MODE_HEAD_QK 1   // bf16 out, [B,H,S,DK]
#define MODE_HEAD_VT 2   // bf16 out, [B,H,DK,S] (transposed V)
#define MODE_F32     3   // fp32 out, row-major [M, N]
#define MODE_BF16_RM 0   // bf16 out, row-major [M, N]

// ---- CDNA5 async global->LDS copy (ASYNCcnt-tracked) ----------------------
// LDS address taken from the vdst VGPR (+ offset); per-lane 16B transfer.
// LDS byte address = low 32 bits of the flat address (ISA 10.2 aperture rule).
__device__ __forceinline__ unsigned ldsOff(const void* p) {
    return (unsigned)(unsigned long long)p;
}
#define ASYNC_LOAD_B128(ldsoff, gaddr)                                   \
    asm volatile("global_load_async_to_lds_b128 %0, %1, off"             \
                 :: "v"(ldsoff), "v"(gaddr) : "memory")
#define WAIT_ASYNCCNT(n)                                                 \
    asm volatile("s_wait_asynccnt " #n ::: "memory")

// ---------------------------------------------------------------------------
// fp32 -> bf16 conversion (vectorized 4/thread)
// ---------------------------------------------------------------------------
__global__ void cvt_bf16_kernel(const float* __restrict__ in,
                                __bf16* __restrict__ out, int n4) {
    int i = blockIdx.x * blockDim.x + threadIdx.x;
    if (i < n4) {
        v4f x = ((const v4f*)in)[i];
        __bf16 y0 = (__bf16)x.x, y1 = (__bf16)x.y, y2 = (__bf16)x.z, y3 = (__bf16)x.w;
        out[i*4+0] = y0; out[i*4+1] = y1; out[i*4+2] = y2; out[i*4+3] = y3;
    }
}

// ---------------------------------------------------------------------------
// GEMM:  C[M x 1024] = A[M x 1024](bf16,row-major) @ W[1024 x 1024]^T + bias
// Block: 256 threads = 8 waves; block tile 128 rows x 64 cols, k-chunk 32.
// The 64x32 W tile (4KB) is shared by all 8 waves -> staged once per block
// per chunk via async-to-LDS (1 b128 per thread), double buffered.
// A-fragments are software-pipelined one chunk ahead in registers.
//   A-frag (16x32): lane L holds row (mBase + L%16);
//       V0-3: K = k0 + 8*(L/16) + 0..7,  V4-7: K = k0 + 16 + 8*(L/16) + 0..7
//   B-frag (32x16): lane L holds W row (nBase + L%16),
//       V0-3: K = k0 + 8*(L/16) + 0..7,  V4-7: K = k0 + 16 + 8*(L/16) + 0..7 (read from LDS)
// ---------------------------------------------------------------------------
__global__ __launch_bounds__(256)
void gemm_bf16_kernel(const __bf16* __restrict__ A,
                      const __bf16* __restrict__ W,
                      const float*  __restrict__ bias,
                      void* __restrict__ out, int mode) {
    __shared__ __align__(16) __bf16 lds_w[2][64][32];   // 8KB double buffer

    const int tid  = threadIdx.x;
    const int wave = tid >> 5;
    const int lane = tid & 31;
    const int hl   = lane >> 4;
    const int ln   = lane & 15;
    const int mBase = blockIdx.y * 128 + wave * 16;
    const int nBase = blockIdx.x * 64;
    const int NCHUNK = DMODEL / 32;

    const __bf16* arow = A + (size_t)(mBase + ln) * DMODEL;

    // staging map: thread -> (row = tid>>2 in [0,64), seg = tid&3 -> 16B)
    const int srow = tid >> 2, sseg = tid & 3;
    const __bf16* wsrc = W + (size_t)(nBase + srow) * DMODEL + sseg * 8;
    unsigned wdst[2] = { ldsOff(&lds_w[0][srow][sseg * 8]),
                         ldsOff(&lds_w[1][srow][sseg * 8]) };

    // prologue: stage chunk 0, preload A-frag chunk 0
    ASYNC_LOAD_B128(wdst[0], (unsigned long long)wsrc);
    ABFrag af;
    af.h[0] = *(const v8bf*)(arow + 8 * hl);
    af.h[1] = *(const v8bf*)(arow + 16 + 8 * hl);

    v8f acc[4] = {v8f{}, v8f{}, v8f{}, v8f{}};

    for (int it = 0; it < NCHUNK; ++it) {
        const int k0 = it * 32;
        ABFrag afn = af;
        if (it + 1 < NCHUNK) {
            // stage next W chunk + pipeline next A fragment
            ASYNC_LOAD_B128(wdst[(it + 1) & 1],
                            (unsigned long long)(wsrc + (size_t)(k0 + 32)));
            afn.h[0] = *(const v8bf*)(arow + k0 + 32 + 8 * hl);
            afn.h[1] = *(const v8bf*)(arow + k0 + 32 + 16 + 8 * hl);
            WAIT_ASYNCCNT(1);      // current chunk landed; next may be in flight
        } else {
            WAIT_ASYNCCNT(0);
        }
        __syncthreads();           // all threads' copies visible

        const int buf = it & 1;
#pragma unroll
        for (int j = 0; j < 4; ++j) {
            ABFrag bf;
            bf.h[0] = *(const v8bf*)(&lds_w[buf][j * 16 + ln][8 * hl]);
            bf.h[1] = *(const v8bf*)(&lds_w[buf][j * 16 + ln][16 + 8 * hl]);
            acc[j] = __builtin_amdgcn_wmma_f32_16x16x32_bf16(
                false, af.v, false, bf.v, (short)0, acc[j], false, false);
        }
        __syncthreads();           // protect buffer overwritten next issue
        af = afn;
    }

    // Epilogue: bias + store per layout mode.
#pragma unroll
    for (int j = 0; j < 4; ++j) {
#pragma unroll
        for (int i = 0; i < 8; ++i) {
            int r = mBase + i + 8 * hl;
            int e = nBase + j * 16 + ln;
            float v = acc[j][i] + bias[e];
            if (mode == MODE_F32) {
                ((float*)out)[(size_t)r * DMODEL + e] = v;
            } else if (mode == MODE_HEAD_QK) {
                int b = r >> 11, s = r & (S_LEN - 1);
                int h = e >> 6, d = e & (DK - 1);
                ((__bf16*)out)[(((size_t)b * NHEAD + h) * S_LEN + s) * DK + d] =
                    (__bf16)v;
            } else if (mode == MODE_HEAD_VT) {
                int b = r >> 11, s = r & (S_LEN - 1);
                int h = e >> 6, d = e & (DK - 1);
                ((__bf16*)out)[(((size_t)b * NHEAD + h) * DK + d) * S_LEN + s] =
                    (__bf16)v;
            } else { // MODE_BF16_RM
                ((__bf16*)out)[(size_t)r * DMODEL + e] = (__bf16)v;
            }
        }
    }
}

// ---------------------------------------------------------------------------
// Flash attention: block = 128 queries of one (b,h); wave = 16-query strip.
// All 8 waves consume the same K/V chunks -> stage them per block via
// async-to-LDS (double buffered): K chunk 32x64 (4KB) + Vt chunk 64x32 (4KB),
// 2 b128 async loads per thread per chunk.
// Per 32-key chunk:
//   S(16x32) = Q(16x64) @ Kc^T          (4 WMMAs, f32 acc)
//   online softmax (row max/sum via 16-lane shfl-xor reductions)
//   P -> bf16 via wave-private LDS slab (C-layout -> A-layout)
//   O(16x64) += P(16x32) @ Vc(32x64)    (4 WMMAs)
// ---------------------------------------------------------------------------
__global__ __launch_bounds__(256)
void flash_attn_kernel(const __bf16* __restrict__ Qh,
                       const __bf16* __restrict__ Kh,
                       const __bf16* __restrict__ Vt,
                       __bf16* __restrict__ At) {
    __shared__ __align__(16) __bf16 lds_k[2][32][64];   // 8KB
    __shared__ __align__(16) __bf16 lds_v[2][64][32];   // 8KB
    __shared__ __align__(16) __bf16 plds[8][16][32];    // 8KB (wave-private)

    const int bh   = blockIdx.y;          // b*NHEAD + h
    const int tid  = threadIdx.x;
    const int wave = tid >> 5;
    const int lane = tid & 31;
    const int hl   = lane >> 4;
    const int ln   = lane & 15;
    const int qBase = blockIdx.x * 128 + wave * 16;
    const int NCHUNK = S_LEN / 32;

    const __bf16* Qp = Qh + ((size_t)bh * S_LEN + qBase) * DK;
    const __bf16* Kp = Kh + (size_t)bh * S_LEN * DK;
    const __bf16* Vp = Vt + (size_t)bh * DK * S_LEN;

    // Q A-fragments for d-chunks [0,32) and [32,64)
    const __bf16* qrow = Qp + (size_t)ln * DK;
    v16bf qf[2];
#pragma unroll
    for (int kc = 0; kc < 2; ++kc) {
        ABFrag t;
        t.h[0] = *(const v8bf*)(qrow + kc * 32 + 8 * hl);
        t.h[1] = *(const v8bf*)(qrow + kc * 32 + 16 + 8 * hl);
        qf[kc] = t.v;
    }

    // staging maps
    const __bf16* ksrc = Kp + (size_t)(tid >> 3) * DK + (tid & 7) * 8;
    const __bf16* vsrc = Vp + (size_t)(tid >> 2) * S_LEN + (tid & 3) * 8;
    unsigned kdst[2] = { ldsOff(&lds_k[0][tid >> 3][(tid & 7) * 8]),
                         ldsOff(&lds_k[1][tid >> 3][(tid & 7) * 8]) };
    unsigned vdst[2] = { ldsOff(&lds_v[0][tid >> 2][(tid & 3) * 8]),
                         ldsOff(&lds_v[1][tid >> 2][(tid & 3) * 8]) };

    // prologue: stage chunk 0
    ASYNC_LOAD_B128(kdst[0], (unsigned long long)ksrc);
    ASYNC_LOAD_B128(vdst[0], (unsigned long long)vsrc);

    float m_i[8], l_i[8];
#pragma unroll
    for (int i = 0; i < 8; ++i) { m_i[i] = -1e30f; l_i[i] = 0.0f; }
    v8f o[4] = {v8f{}, v8f{}, v8f{}, v8f{}};

    const float scale = 0.125f;  // 1/sqrt(DK)

    for (int it = 0; it < NCHUNK; ++it) {
        const int key0 = it * 32;
        if (it + 1 < NCHUNK) {
            ASYNC_LOAD_B128(kdst[(it + 1) & 1],
                            (unsigned long long)(ksrc + (size_t)(key0 + 32) * DK));
            ASYNC_LOAD_B128(vdst[(it + 1) & 1],
                            (unsigned long long)(vsrc + (key0 + 32)));
            WAIT_ASYNCCNT(2);   // current chunk's 2 copies landed (in-order)
        } else {
            WAIT_ASYNCCNT(0);
        }
        __syncthreads();
        const int buf = it & 1;

        // ---- scores: two 16-column tiles (B-frags from LDS) ------------
        v8f s[2];
#pragma unroll
        for (int t = 0; t < 2; ++t) {
            v8f a = {};
#pragma unroll
            for (int kc = 0; kc < 2; ++kc) {
                ABFrag bf;
                bf.h[0] = *(const v8bf*)(&lds_k[buf][t * 16 + ln][kc * 32 + 8 * hl]);
                bf.h[1] = *(const v8bf*)(&lds_k[buf][t * 16 + ln][kc * 32 + 16 + 8 * hl]);
                a = __builtin_amdgcn_wmma_f32_16x16x32_bf16(
                    false, qf[kc], false, bf.v, (short)0, a, false, false);
            }
            s[t] = a;
        }

        // ---- online softmax (per row i; 16 columns live in 16 lanes) ---
        float alpha[8];
#pragma unroll
        for (int i = 0; i < 8; ++i) {
            float s0 = s[0][i] * scale;
            float s1 = s[1][i] * scale;
            float mx = fmaxf(s0, s1);
#pragma unroll
            for (int off = 8; off >= 1; off >>= 1)
                mx = fmaxf(mx, __shfl_xor(mx, off, 16));
            float mnew = fmaxf(m_i[i], mx);
            float p0 = __expf(s0 - mnew);
            float p1 = __expf(s1 - mnew);
            float rs = p0 + p1;
#pragma unroll
            for (int off = 8; off >= 1; off >>= 1)
                rs += __shfl_xor(rs, off, 16);
            alpha[i] = __expf(m_i[i] - mnew);
            l_i[i]   = l_i[i] * alpha[i] + rs;
            m_i[i]   = mnew;
            plds[wave][i + 8 * hl][ln]      = (__bf16)p0;
            plds[wave][i + 8 * hl][16 + ln] = (__bf16)p1;
        }

        // rescale running output
#pragma unroll
        for (int j = 0; j < 4; ++j)
#pragma unroll
            for (int i = 0; i < 8; ++i) o[j][i] *= alpha[i];

        // ---- reload P as 16x32 A-fragment (same-wave LDS, in-order) ----
        ABFrag pa;
        pa.h[0] = *(const v8bf*)(&plds[wave][ln][8 * hl]);
        pa.h[1] = *(const v8bf*)(&plds[wave][ln][16 + 8 * hl]);

        // ---- O += P @ Vc  (B-frags from LDS) ---------------------------
#pragma unroll
        for (int j = 0; j < 4; ++j) {
            ABFrag bf;
            bf.h[0] = *(const v8bf*)(&lds_v[buf][j * 16 + ln][8 * hl]);
            bf.h[1] = *(const v8bf*)(&lds_v[buf][j * 16 + ln][16 + 8 * hl]);
            o[j] = __builtin_amdgcn_wmma_f32_16x16x32_bf16(
                false, pa.v, false, bf.v, (short)0, o[j], false, false);
        }
        __syncthreads();    // protect buffer overwritten by next issue
    }

    // ---- normalize + store bf16 [B*S, H*DK] row-major ------------------
    const int b = bh >> 4, h = bh & (NHEAD - 1);
#pragma unroll
    for (int j = 0; j < 4; ++j) {
#pragma unroll
        for (int i = 0; i < 8; ++i) {
            int row = qBase + i + 8 * hl;
            int col = h * DK + j * 16 + ln;
            float v = o[j][i] / l_i[i];
            At[((size_t)b * S_LEN + row) * DMODEL + col] = (__bf16)v;
        }
    }
}

// ---------------------------------------------------------------------------
// Host-side launch
// ---------------------------------------------------------------------------
extern "C" void kernel_launch(void* const* d_in, const int* in_sizes, int n_in,
                              void* d_out, int out_size, void* d_ws, size_t ws_size,
                              hipStream_t stream) {
    const float* x_q = (const float*)d_in[0];
    const float* x_k = (const float*)d_in[1];
    const float* x_v = (const float*)d_in[2];
    const float* Wq  = (const float*)d_in[3];
    const float* bq  = (const float*)d_in[4];
    const float* Wk  = (const float*)d_in[5];
    const float* bk  = (const float*)d_in[6];
    const float* Wv  = (const float*)d_in[7];
    const float* bv  = (const float*)d_in[8];
    const float* Wo  = (const float*)d_in[9];
    const float* bo  = (const float*)d_in[10];

    const size_t NX = (size_t)MROWS * DMODEL;   // 8M elems
    const size_t NW = (size_t)DMODEL * DMODEL;  // 1M elems

    char* ws = (char*)d_ws;
    __bf16* Xq  = (__bf16*)(ws);                          // 16 MB
    __bf16* Xk  = (__bf16*)(ws + 2 * NX);                 // 16 MB
    __bf16* Xv  = (__bf16*)(ws + 4 * NX);                 // 16 MB
    __bf16* Wqb = (__bf16*)(ws + 6 * NX);                 //  2 MB
    __bf16* Wkb = (__bf16*)(ws + 6 * NX + 2 * NW);
    __bf16* Wvb = (__bf16*)(ws + 6 * NX + 4 * NW);
    __bf16* Wob = (__bf16*)(ws + 6 * NX + 6 * NW);
    __bf16* Qh  = (__bf16*)(ws + 6 * NX + 8 * NW);        // 16 MB [B,H,S,DK]
    __bf16* Kh  = (__bf16*)(ws + 8 * NX + 8 * NW);        // 16 MB [B,H,S,DK]
    __bf16* Vt  = (__bf16*)(ws + 10 * NX + 8 * NW);       // 16 MB [B,H,DK,S]
    __bf16* At  = Xq;  // attention output aliases Xq (dead after Q-projection)

    // 1) fp32 -> bf16
    {
        int n4x = (int)(NX / 4), n4w = (int)(NW / 4);
        int tb = 256;
        cvt_bf16_kernel<<<(n4x + tb - 1) / tb, tb, 0, stream>>>(x_q, Xq, n4x);
        cvt_bf16_kernel<<<(n4x + tb - 1) / tb, tb, 0, stream>>>(x_k, Xk, n4x);
        cvt_bf16_kernel<<<(n4x + tb - 1) / tb, tb, 0, stream>>>(x_v, Xv, n4x);
        cvt_bf16_kernel<<<(n4w + tb - 1) / tb, tb, 0, stream>>>(Wq, Wqb, n4w);
        cvt_bf16_kernel<<<(n4w + tb - 1) / tb, tb, 0, stream>>>(Wk, Wkb, n4w);
        cvt_bf16_kernel<<<(n4w + tb - 1) / tb, tb, 0, stream>>>(Wv, Wvb, n4w);
        cvt_bf16_kernel<<<(n4w + tb - 1) / tb, tb, 0, stream>>>(Wo, Wob, n4w);
    }

    // 2) projections (M=8192 rows; grid = N/64 x M/128)
    dim3 ggrid(DMODEL / 64, MROWS / 128);
    gemm_bf16_kernel<<<ggrid, 256, 0, stream>>>(Xq, Wqb, bq, Qh, MODE_HEAD_QK);
    gemm_bf16_kernel<<<ggrid, 256, 0, stream>>>(Xk, Wkb, bk, Kh, MODE_HEAD_QK);
    gemm_bf16_kernel<<<ggrid, 256, 0, stream>>>(Xv, Wvb, bv, Vt, MODE_HEAD_VT);

    // 3) flash attention: grid = (S/128 q-tiles, B*H)
    dim3 agrid(S_LEN / 128, NBATCH * NHEAD);
    flash_attn_kernel<<<agrid, 256, 0, stream>>>(Qh, Kh, Vt, At);

    // 4) output projection -> fp32 d_out
    gemm_bf16_kernel<<<ggrid, 256, 0, stream>>>(At, Wob, bo, d_out, MODE_F32);
}
